// FoveatedEncoderWithAblations_47811575939229
// MI455X (gfx1250) — matrix operand order repaired and checked by reading, not compiled, not run on hardware
//
#include <hip/hip_runtime.h>

typedef __attribute__((ext_vector_type(16))) _Float16 v16h;
typedef __attribute__((ext_vector_type(8)))  float    v8f;

#define B_   16
#define N_   1370
#define D_   384
#define H_   6
#define HD_  64
#define DFF_ 1536
#define L_   12

// ---------------------------------------------------------------------------
// WMMA tile helper: one wave computes a 16x16 tile of Y = X[16,kdim] @ W[kdim,:]
// X, W are f32 (generic pointers: LDS or global); converted to f16 fragments.
// A layout (16-bit 16x32): lanes 0-15 row M=lane, halfs0-7=K k0..k0+7,
// halfs8-15=K k0+16..k0+23; lanes 16-31 row M=lane-16, K offsets +8 / +24.
// B layout (16-bit 32x16): lanes 0-15 col=lane halfs0-7=K k0..k0+7, halfs8-15=K k0+16..k0+23; lanes16-31 K +8 / +24.
// ---------------------------------------------------------------------------
__device__ inline v8f gemm_tile(const float* X, int ldx, const float* W, int ldw,
                                int n0, int kdim, int lane) {
  v8f c = {};
  const int m   = lane & 15;
  const int kh  = (lane >> 4) & 1;
  const int col = lane & 15;
  for (int k0 = 0; k0 < kdim; k0 += 32) {
    v16h a, b;
    const float* xrow = X + m * ldx + k0 + kh * 8;
#pragma unroll
    for (int i = 0; i < 8; ++i) {
      a[i]     = (_Float16)xrow[i];
      a[8 + i] = (_Float16)xrow[16 + i];
    }
    const float* wcol = W + (k0 + kh * 8) * ldw + n0 + col;
#pragma unroll
    for (int j = 0; j < 16; ++j) b[j] = (_Float16)wcol[(j + (j >= 8 ? 8 : 0)) * ldw];
    c = __builtin_amdgcn_wmma_f32_16x16x32_f16(false, a, false, b, (short)0, c,
                                               false, false);
  }
  return c;
}

// D/C tile layout: lanes 0-15: col N=lane, VGPR r -> row M=r; lanes 16-31:
// col N=lane-16, VGPR r -> row M=r+8.
__device__ inline int tile_mbase(int lane) { return (lane >= 16) ? 8 : 0; }

// LayerNorm of a [16, 384] matrix, 256 threads (8 waves, 2 rows each).
__device__ inline void ln16(const float* x, float* y, const float* g,
                            const float* bt) {
  int wave = threadIdx.x >> 5, lane = threadIdx.x & 31;
  for (int r = wave; r < B_; r += 8) {
    float s = 0.f, s2 = 0.f;
    for (int i = lane; i < D_; i += 32) {
      float v = x[r * D_ + i];
      s += v; s2 += v * v;
    }
#pragma unroll
    for (int off = 16; off; off >>= 1) {
      s  += __shfl_xor(s,  off, 32);
      s2 += __shfl_xor(s2, off, 32);
    }
    float mean = s * (1.0f / D_);
    float var  = s2 * (1.0f / D_) - mean * mean;
    float inv  = rsqrtf(var + 1e-6f);
    for (int i = lane; i < D_; i += 32)
      y[r * D_ + i] = (x[r * D_ + i] - mean) * inv * g[i] + bt[i];
  }
}

// ---------------------------------------------------------------------------
// q0 = query @ w_in   (single block, 24 tiles over 8 waves)
// ---------------------------------------------------------------------------
__global__ void k_in_proj(const float* __restrict__ query,
                          const float* __restrict__ w_in, float* __restrict__ q) {
  int wave = threadIdx.x >> 5, lane = threadIdx.x & 31;
  for (int t = wave; t < 24; t += 8) {
    v8f c = gemm_tile(query, D_, w_in, D_, t * 16, D_, lane);
    int col = lane & 15, mb = tile_mbase(lane);
#pragma unroll
    for (int r = 0; r < 8; ++r) q[(mb + r) * D_ + t * 16 + col] = c[r];
  }
}

// ---------------------------------------------------------------------------
// Per-layer pre: qn = LN(q); Qs = (qn@Wq + bq)/scale; pack Qmask B-fragments.
// qfrag layout: [b][chunk(12)][lane(32)][16 halfs], directly loadable as v16h.
// ---------------------------------------------------------------------------
__global__ void k_layer_pre(const float* __restrict__ q,
                            const float* __restrict__ g,
                            const float* __restrict__ bt,
                            const float* __restrict__ Wq,
                            const float* __restrict__ bq,
                            float* __restrict__ Qs,
                            _Float16* __restrict__ qfrag) {
  __shared__ float qn[B_ * D_];
  ln16(q, qn, g, bt);
  __syncthreads();
  int wave = threadIdx.x >> 5, lane = threadIdx.x & 31;
  const float scale_inv = 0.125f;  // 1/sqrt(64)
  for (int t = wave; t < 24; t += 8) {
    v8f c = gemm_tile(qn, D_, Wq, D_, t * 16, D_, lane);
    int col = lane & 15, mb = tile_mbase(lane);
#pragma unroll
    for (int r = 0; r < 8; ++r)
      Qs[(mb + r) * D_ + t * 16 + col] = (c[r] + bq[t * 16 + col]) * scale_inv;
  }
  __syncthreads();
  // Build Qmask fragments: B[k_local, c] = Qs[b][k] if c == head(k) else 0
  for (int t = threadIdx.x; t < B_ * 12 * 32; t += 256) {
    int lane_s = t & 31;
    int kc = (t >> 5) % 12;
    int b  = t / (12 * 32);
    int col = lane_s & 15, kh = lane_s >> 4;
    _Float16* dst = qfrag + (size_t)t * 16;
#pragma unroll
    for (int j = 0; j < 16; ++j) {
      int k = kc * 32 + kh * 8 + j + (j >= 8 ? 8 : 0);
      int head = k >> 6;
      dst[j] = (col == head) ? (_Float16)Qs[b * D_ + k] : (_Float16)0.0f;
    }
  }
}

// ---------------------------------------------------------------------------
// Streaming score kernel: the bandwidth-critical one.  Each wave owns one
// 16-row n-tile of one batch; 12 chained WMMAs over K=384, A from global K
// (float4 loads), B = preloaded Qmask fragment.  S[b*6+h][n].
// ---------------------------------------------------------------------------
__global__ void k_score(const float* __restrict__ Kg,
                        const _Float16* __restrict__ qfrag,
                        float* __restrict__ S) {
  int wave = threadIdx.x >> 5, lane = threadIdx.x & 31;
  int tile = blockIdx.x * 8 + wave;          // 86 tiles of 16 rows
  if (tile >= (N_ + 15) / 16) return;        // uniform per wave
  int b  = blockIdx.y;
  int n0 = tile * 16;
  int m  = lane & 15;
  int row = n0 + m; if (row >= N_) row = N_ - 1;   // clamp (dup reads ok)
  int kh = lane >> 4;
  const float* Kb = Kg + (size_t)b * N_ * D_;
  const _Float16* qf = qfrag + ((size_t)(b * 12) * 32 + lane) * 16;
  v8f c = {};
#pragma unroll
  for (int kc = 0; kc < 12; ++kc) {
    const float* xrow = Kb + (size_t)row * D_ + kc * 32 + kh * 8;
    float4 f0 = *(const float4*)(xrow);
    float4 f1 = *(const float4*)(xrow + 4);
    float4 f2 = *(const float4*)(xrow + 16);
    float4 f3 = *(const float4*)(xrow + 20);
    v16h a;
    a[0]=(_Float16)f0.x; a[1]=(_Float16)f0.y; a[2]=(_Float16)f0.z; a[3]=(_Float16)f0.w;
    a[4]=(_Float16)f1.x; a[5]=(_Float16)f1.y; a[6]=(_Float16)f1.z; a[7]=(_Float16)f1.w;
    a[8]=(_Float16)f2.x; a[9]=(_Float16)f2.y; a[10]=(_Float16)f2.z; a[11]=(_Float16)f2.w;
    a[12]=(_Float16)f3.x; a[13]=(_Float16)f3.y; a[14]=(_Float16)f3.z; a[15]=(_Float16)f3.w;
    v16h bfrag = *(const v16h*)(qf + (size_t)kc * 32 * 16);
    c = __builtin_amdgcn_wmma_f32_16x16x32_f16(false, a, false, bfrag, (short)0,
                                               c, false, false);
  }
  int col = lane & 15;
  if (col < H_) {
    int mb = tile_mbase(lane);
#pragma unroll
    for (int r = 0; r < 8; ++r) {
      int n = n0 + mb + r;
      if (n < N_) S[((size_t)b * H_ + col) * N_ + n] = c[r];
    }
  }
}

// ---------------------------------------------------------------------------
// Top-k + softmax + sparse V gather.  One block per (b,h).  Only the 16
// selected V rows are read -> V is never streamed.
// ---------------------------------------------------------------------------
__global__ void k_topk_attn(const float* __restrict__ S,
                            const float* __restrict__ Vg,
                            float* __restrict__ attn,
                            const int* __restrict__ ktp) {
  int bh = blockIdx.x, b = bh / H_, h = bh % H_;
  int kk = *ktp;
  if (kk > 64) kk = 64;
  if (kk < 1)  kk = 1;
  __shared__ float s_ls[1376];
  __shared__ float red_v[256];
  __shared__ int   red_i[256];
  __shared__ int   top_idx[64];
  __shared__ float top_val[64];
  __shared__ float w_sh[64];
  const float* Sp = S + (size_t)bh * N_;
  for (int i = threadIdx.x; i < 1376; i += 256)
    s_ls[i] = (i < N_) ? Sp[i] : -INFINITY;
  __syncthreads();
  for (int t = 0; t < kk; ++t) {
    float best = -INFINITY; int bi = -1;
    for (int i = threadIdx.x; i < N_; i += 256) {
      float v = s_ls[i];
      if (v > best) { best = v; bi = i; }
    }
    red_v[threadIdx.x] = best; red_i[threadIdx.x] = bi;
    __syncthreads();
    for (int off = 128; off > 0; off >>= 1) {
      if (threadIdx.x < off) {
        float ov = red_v[threadIdx.x + off]; int oi = red_i[threadIdx.x + off];
        float mv = red_v[threadIdx.x];       int mi = red_i[threadIdx.x];
        if (ov > mv || (ov == mv && oi >= 0 && (mi < 0 || oi < mi))) {
          red_v[threadIdx.x] = ov; red_i[threadIdx.x] = oi;
        }
      }
      __syncthreads();
    }
    if (threadIdx.x == 0) {
      int wi = red_i[0];
      top_idx[t] = (wi < 0) ? 0 : wi;
      top_val[t] = red_v[0];
      if (wi >= 0) s_ls[wi] = -INFINITY;
    }
    __syncthreads();
  }
  if (threadIdx.x == 0) {
    float mx = top_val[0];                 // non-increasing -> [0] is max
    float sum = 0.f;
    for (int t = 0; t < kk; ++t) { float e = __expf(top_val[t] - mx); w_sh[t] = e; sum += e; }
    float inv = 1.0f / sum;
    for (int t = 0; t < kk; ++t) w_sh[t] *= inv;
  }
  __syncthreads();
  if (threadIdx.x < HD_) {
    int d = threadIdx.x;
    float acc = 0.f;
    for (int t = 0; t < kk; ++t)
      acc += w_sh[t] * Vg[((size_t)b * N_ + top_idx[t]) * D_ + h * HD_ + d];
    attn[b * D_ + h * HD_ + d] = acc;
  }
}

// ---------------------------------------------------------------------------
// y = attn@Wo + bo;  q += y*ls1;  qn2 = LN(q) with norm2 params.
// ---------------------------------------------------------------------------
__global__ void k_attn_out(const float* __restrict__ attn,
                           const float* __restrict__ Wo,
                           const float* __restrict__ bo,
                           const float* __restrict__ ls1,
                           const float* __restrict__ n2g,
                           const float* __restrict__ n2b,
                           float* __restrict__ q, float* __restrict__ qn2) {
  __shared__ float y[B_ * D_];
  int wave = threadIdx.x >> 5, lane = threadIdx.x & 31;
  for (int t = wave; t < 24; t += 8) {
    v8f c = gemm_tile(attn, D_, Wo, D_, t * 16, D_, lane);
    int col = lane & 15, mb = tile_mbase(lane);
#pragma unroll
    for (int r = 0; r < 8; ++r) y[(mb + r) * D_ + t * 16 + col] = c[r];
  }
  __syncthreads();
  for (int i = threadIdx.x; i < B_ * D_; i += 256) {
    int col = i % D_;
    q[i] = q[i] + (y[i] + bo[col]) * ls1[col];
  }
  __syncthreads();
  ln16(q, qn2, n2g, n2b);
}

// ---------------------------------------------------------------------------
// h1 = gelu(qn2 @ W1 + b1)   96 tiles = 12 blocks x 8 waves.
// ---------------------------------------------------------------------------
__global__ void k_ffn1(const float* __restrict__ qn2,
                       const float* __restrict__ W1,
                       const float* __restrict__ b1, float* __restrict__ h1) {
  int wave = threadIdx.x >> 5, lane = threadIdx.x & 31;
  int tile = blockIdx.x * 8 + wave;
  int n0 = tile * 16;
  v8f c = gemm_tile(qn2, D_, W1, DFF_, n0, D_, lane);
  int col = lane & 15, mb = tile_mbase(lane);
#pragma unroll
  for (int r = 0; r < 8; ++r) {
    float x = c[r] + b1[n0 + col];
    float gv = 0.5f * x * (1.0f + erff(x * 0.70710678118654752f));
    h1[(mb + r) * DFF_ + n0 + col] = gv;
  }
}

// ---------------------------------------------------------------------------
// q += (h1 @ W2 + b2) * ls2
// ---------------------------------------------------------------------------
__global__ void k_ffn2(const float* __restrict__ h1,
                       const float* __restrict__ W2,
                       const float* __restrict__ b2,
                       const float* __restrict__ ls2, float* __restrict__ q) {
  int wave = threadIdx.x >> 5, lane = threadIdx.x & 31;
  for (int t = wave; t < 24; t += 8) {
    v8f c = gemm_tile(h1, DFF_, W2, D_, t * 16, DFF_, lane);
    int col = lane & 15, mb = tile_mbase(lane);
#pragma unroll
    for (int r = 0; r < 8; ++r) {
      int idx = (mb + r) * D_ + t * 16 + col;
      q[idx] = q[idx] + (c[r] + b2[t * 16 + col]) * ls2[t * 16 + col];
    }
  }
}

// ---------------------------------------------------------------------------
// out = LN(q) @ w_out + b_out
// ---------------------------------------------------------------------------
__global__ void k_final(const float* __restrict__ q,
                        const float* __restrict__ lng,
                        const float* __restrict__ lnb,
                        const float* __restrict__ w_out,
                        const float* __restrict__ b_out,
                        float* __restrict__ out) {
  __shared__ float qn[B_ * D_];
  ln16(q, qn, lng, lnb);
  __syncthreads();
  int wave = threadIdx.x >> 5, lane = threadIdx.x & 31;
  for (int t = wave; t < 24; t += 8) {
    v8f c = gemm_tile(qn, D_, w_out, D_, t * 16, D_, lane);
    int col = lane & 15, mb = tile_mbase(lane);
#pragma unroll
    for (int r = 0; r < 8; ++r)
      out[(mb + r) * D_ + t * 16 + col] = c[r] + b_out[t * 16 + col];
  }
}

// ---------------------------------------------------------------------------
extern "C" void kernel_launch(void* const* d_in, const int* in_sizes, int n_in,
                              void* d_out, int out_size, void* d_ws,
                              size_t ws_size, hipStream_t stream) {
  (void)in_sizes; (void)n_in; (void)out_size; (void)ws_size;
  const float* query = (const float*)d_in[0];
  const float* Kt    = (const float*)d_in[1];
  const float* Vt    = (const float*)d_in[2];
  const float* w_in  = (const float*)d_in[3];
  const float* n1g   = (const float*)d_in[4];
  const float* n1b   = (const float*)d_in[5];
  const float* Wq    = (const float*)d_in[6];
  const float* bq    = (const float*)d_in[7];
  const float* Wo    = (const float*)d_in[8];
  const float* bo    = (const float*)d_in[9];
  const float* ls1   = (const float*)d_in[10];
  const float* n2g   = (const float*)d_in[11];
  const float* n2b   = (const float*)d_in[12];
  const float* W1    = (const float*)d_in[13];
  const float* b1    = (const float*)d_in[14];
  const float* W2    = (const float*)d_in[15];
  const float* b2    = (const float*)d_in[16];
  const float* ls2   = (const float*)d_in[17];
  const float* lng   = (const float*)d_in[18];
  const float* lnb   = (const float*)d_in[19];
  const float* w_out = (const float*)d_in[20];
  const float* b_out = (const float*)d_in[21];
  const int*   ktop  = (const int*)d_in[22];

  char* ws = (char*)d_ws;
  float*    q_ws  = (float*)(ws + 0);        // 24576 B
  float*    Qs    = (float*)(ws + 24576);    // 24576 B
  float*    qn2   = (float*)(ws + 49152);    // 24576 B
  float*    attn  = (float*)(ws + 73728);    // 24576 B
  float*    h1    = (float*)(ws + 98304);    // 98304 B
  float*    S     = (float*)(ws + 196608);   // 526080 B
  _Float16* qfrag = (_Float16*)(ws + 722688); // 196608 B

  const size_t kv_stride = (size_t)B_ * N_ * D_;
  const size_t w_stride  = (size_t)D_ * D_;
  const size_t w1_stride = (size_t)D_ * DFF_;

  k_in_proj<<<1, 256, 0, stream>>>(query, w_in, q_ws);
  for (int l = 0; l < L_; ++l) {
    const float* Kl = Kt + (size_t)l * kv_stride;
    const float* Vl = Vt + (size_t)l * kv_stride;
    k_layer_pre<<<1, 256, 0, stream>>>(q_ws, n1g + l * D_, n1b + l * D_,
                                       Wq + l * w_stride, bq + l * D_, Qs, qfrag);
    k_score<<<dim3(11, B_), 256, 0, stream>>>(Kl, qfrag, S);
    k_topk_attn<<<B_ * H_, 256, 0, stream>>>(S, Vl, attn, ktop);
    k_attn_out<<<1, 256, 0, stream>>>(attn, Wo + l * w_stride, bo + l * D_,
                                      ls1 + l * D_, n2g + l * D_, n2b + l * D_,
                                      q_ws, qn2);
    k_ffn1<<<12, 256, 0, stream>>>(qn2, W1 + l * w1_stride, b1 + l * DFF_, h1);
    k_ffn2<<<1, 256, 0, stream>>>(h1, W2 + l * w1_stride, b2 + l * D_,
                                  ls2 + l * D_, q_ws);
  }
  k_final<<<1, 256, 0, stream>>>(q_ws, lng, lnb, w_out, b_out, (float*)d_out);
}
